// GraphEdgeAttenNetworkLayers_masking_13305808683262
// MI455X (gfx1250) — compile-verified, matrix operand order checked by reading, not run on hardware
//
#include <hip/hip_runtime.h>
#include <hip/hip_bf16.h>

// ---------------------------------------------------------------------------
// GraphEdgeAttenNetwork (2 layers) for gfx1250 (MI455X).
// GEMMs on v_wmma_f32_16x16x32_bf16. Weights repacked on-device into per-lane
// WMMA B-fragment layout (one 32B load per fragment). K-loop double-buffered
// so WMMAs overlap fragment loads. q/k projections store transposed per-edge
// so the fused attention kernel loads its A fragments as aligned b128 vectors.
// Softmax via wave32 shfl; scatter-max via ordered-uint global atomics.
// ---------------------------------------------------------------------------

typedef __bf16 bf16;
typedef __attribute__((ext_vector_type(16))) __bf16 v16bf;
typedef __attribute__((ext_vector_type(8)))  __bf16 v8bf;
typedef __attribute__((ext_vector_type(8)))  float  v8f;

#define DEV static __device__ __forceinline__

static constexpr long long NN = 20000;   // nodes
static constexpr long long EE = 200000;  // edges

DEV v8f wmma_bf16(v16bf a, v16bf b, v8f c) {
  return __builtin_amdgcn_wmma_f32_16x16x32_bf16(
      /*neg_a=*/false, a, /*neg_b=*/false, b,
      /*c_mod=*/(short)0, c, /*reuse_a=*/false, /*reuse_b=*/false);
}

// Packed-B fragment: Wp layout is [ktile][ntile][lane][16] contiguous bf16.
DEV v16bf load_b_packed(const bf16* __restrict__ Wp, int ktile, int ntile,
                        int ntiles, int lane) {
  const bf16* p = Wp + (((size_t)ktile * ntiles + ntile) * 32 + lane) * 16;
  return *(const v16bf*)p;
}

// A fragment from one fp32 row (two contiguous 8-float runs, native cvt).
DEV v16bf load_a_f32(const float* __restrict__ row, int kofs, int half) {
  const float* s0 = row + kofs + half * 8;
  v16bf a;
#pragma unroll
  for (int i = 0; i < 8; ++i) {
    a[i] = (bf16)s0[i];
    a[i + 8] = (bf16)s0[16 + i];
  }
  return a;
}

// A fragment from one bf16 row: two 16B vector loads + register concat.
DEV v16bf load_a_bf16(const bf16* __restrict__ row, int kofs, int half) {
  v8bf lo = *(const v8bf*)(row + kofs + half * 8);
  v8bf hi = *(const v8bf*)(row + kofs + half * 8 + 16);
  return __builtin_shufflevector(lo, hi, 0, 1, 2, 3, 4, 5, 6, 7,
                                 8, 9, 10, 11, 12, 13, 14, 15);
}

// ---------------------------------------------------------------------------
// Repack fp32 W[K x N] (row-major) into bf16 fragment-ready layout.
//   half = lane>>4; k = kt*32 + (i<8 ? half*8+i : 16+half*8+(i-8));
//   n = nt*16 + (lane&15).
// ---------------------------------------------------------------------------
__global__ __launch_bounds__(256) void repack_w_kernel(
    const float* __restrict__ W, bf16* __restrict__ Wp, int K, int N) {
  const int d = blockIdx.x * 256 + threadIdx.x;
  if (d >= K * N) return;
  const int i = d & 15;
  const int lane = (d >> 4) & 31;
  const int tile = d >> 9;
  const int ntiles = N >> 4;
  const int kt = tile / ntiles, nt = tile % ntiles;
  const int half = lane >> 4;
  const int n = nt * 16 + (lane & 15);
  const int k = kt * 32 + ((i < 8) ? (half * 8 + i) : (16 + half * 8 + (i - 8)));
  Wp[d] = (bf16)W[(size_t)k * N + n];
}

// Output modes for the gather-GEMM epilogue.
enum { OM_F32 = 0, OM_BF16 = 1, OM_QKT = 2 };  // QKT: out[e][h][c], n = c*8+h

// ---------------------------------------------------------------------------
// Generic WMMA GEMM with fused up-to-3-segment row gather on A (fp32 in).
// Each wave computes a 16x64 output strip; K-loop double-buffered.
// ---------------------------------------------------------------------------
template <int OUT_MODE, bool RELU>
__global__ __launch_bounds__(256) void gemm_a3_kernel(
    const float* __restrict__ p0, const int* __restrict__ i0, int w0,
    const float* __restrict__ p1, const int* __restrict__ i1, int w1,
    const float* __restrict__ p2, const int* __restrict__ i2, int w2,
    int Mtiles, int N, int K,
    const bf16* __restrict__ Wp, const float* __restrict__ bias,
    float* __restrict__ outF, bf16* __restrict__ outB) {
  const int gw   = (int)((blockIdx.x * blockDim.x + threadIdx.x) >> 5);
  const int lane = threadIdx.x & 31;
  const int ntg  = N >> 6;  // groups of 64 columns
  const int ntiles = N >> 4;
  const int mt = gw / ntg, ng = gw % ntg;
  if (mt >= Mtiles) return;
  const int mrow = mt * 16 + (lane & 15);
  const int half = lane >> 4;
  const float* r0 = p0 ? p0 + (size_t)(i0 ? i0[mrow] : mrow) * w0 : nullptr;
  const float* r1 = p1 ? p1 + (size_t)(i1 ? i1[mrow] : mrow) * w1 : nullptr;
  const float* r2 = p2 ? p2 + (size_t)(i2 ? i2[mrow] : mrow) * w2 : nullptr;

  auto loadA = [&](int kb) -> v16bf {
    const float* src;
    int ko;
    if (kb < w0)           { src = r0; ko = kb; }
    else if (kb < w0 + w1) { src = r1; ko = kb - w0; }
    else                   { src = r2; ko = kb - w0 - w1; }
    return load_a_f32(src, ko, half);
  };

  const v8f cz = {0.f, 0.f, 0.f, 0.f, 0.f, 0.f, 0.f, 0.f};
  v8f c[4] = {cz, cz, cz, cz};

  const int KT = K >> 5;
  v16bf a = loadA(0);
  v16bf b[4];
#pragma unroll
  for (int j = 0; j < 4; ++j) b[j] = load_b_packed(Wp, 0, ng * 4 + j, ntiles, lane);

  for (int kt = 1; kt < KT; ++kt) {          // double-buffered main loop
    v16bf an = loadA(kt * 32);
    v16bf bn[4];
#pragma unroll
    for (int j = 0; j < 4; ++j)
      bn[j] = load_b_packed(Wp, kt, ng * 4 + j, ntiles, lane);
#pragma unroll
    for (int j = 0; j < 4; ++j) c[j] = wmma_bf16(a, b[j], c[j]);
    a = an;
#pragma unroll
    for (int j = 0; j < 4; ++j) b[j] = bn[j];
  }
#pragma unroll
  for (int j = 0; j < 4; ++j) c[j] = wmma_bf16(a, b[j], c[j]);

#pragma unroll
  for (int j = 0; j < 4; ++j) {
    const int n = (ng * 4 + j) * 16 + (lane & 15);
    const float bv = bias[n];
#pragma unroll
    for (int r = 0; r < 8; ++r) {
      const int m = mt * 16 + r + half * 8;  // C/D layout: VGPR r -> M=r / r+8
      float v = c[j][r] + bv;
      if (RELU) v = fmaxf(v, 0.f);
      if (OUT_MODE == OM_F32)  outF[(size_t)m * N + n] = v;
      if (OUT_MODE == OM_BF16) outB[(size_t)m * N + n] = (bf16)v;
      if (OUT_MODE == OM_QKT)  // n = c*8+h -> out[e*256 + h*32 + c]
        outB[(size_t)m * 256 + (n & 7) * 32 + (n >> 3)] = (bf16)v;
    }
  }
}

// GEMM with contiguous bf16 A (second layer of the MLPs). fp32 out.
template <bool RELU>
__global__ __launch_bounds__(256) void gemm_bf16a_kernel(
    const bf16* __restrict__ A, int Mtiles, int N, int K,
    const bf16* __restrict__ Wp, const float* __restrict__ bias,
    float* __restrict__ outF) {
  const int gw   = (int)((blockIdx.x * blockDim.x + threadIdx.x) >> 5);
  const int lane = threadIdx.x & 31;
  const int ntg  = N >> 6;
  const int ntiles = N >> 4;
  const int mt = gw / ntg, ng = gw % ntg;
  if (mt >= Mtiles) return;
  const int half = lane >> 4;
  const bf16* row = A + (size_t)(mt * 16 + (lane & 15)) * K;

  const v8f cz = {0.f, 0.f, 0.f, 0.f, 0.f, 0.f, 0.f, 0.f};
  v8f c[4] = {cz, cz, cz, cz};

  const int KT = K >> 5;
  v16bf a = load_a_bf16(row, 0, half);
  v16bf b[4];
#pragma unroll
  for (int j = 0; j < 4; ++j) b[j] = load_b_packed(Wp, 0, ng * 4 + j, ntiles, lane);

  for (int kt = 1; kt < KT; ++kt) {          // double-buffered main loop
    v16bf an = load_a_bf16(row, kt * 32, half);
    v16bf bn[4];
#pragma unroll
    for (int j = 0; j < 4; ++j)
      bn[j] = load_b_packed(Wp, kt, ng * 4 + j, ntiles, lane);
#pragma unroll
    for (int j = 0; j < 4; ++j) c[j] = wmma_bf16(a, b[j], c[j]);
    a = an;
#pragma unroll
    for (int j = 0; j < 4; ++j) b[j] = bn[j];
  }
#pragma unroll
  for (int j = 0; j < 4; ++j) c[j] = wmma_bf16(a, b[j], c[j]);

#pragma unroll
  for (int j = 0; j < 4; ++j) {
    const int n = (ng * 4 + j) * 16 + (lane & 15);
    const float bv = bias[n];
#pragma unroll
    for (int r = 0; r < 8; ++r) {
      const int m = mt * 16 + r + half * 8;
      float v = c[j][r] + bv;
      if (RELU) v = fmaxf(v, 0.f);
      outF[(size_t)m * N + n] = v;
    }
  }
}

// ---------------------------------------------------------------------------
// Ordered-uint encoding of float so unsigned max == float max.
// ---------------------------------------------------------------------------
DEV unsigned encf(float f) {
  unsigned u = __float_as_uint(f);
  return (u & 0x80000000u) ? ~u : (u | 0x80000000u);
}

// ---------------------------------------------------------------------------
// Fused attention over the virtual [E*8, 64] matrix (row = e*8+h).
// q/k are stored transposed per edge (qT[e][h][c]) so each A fragment is two
// aligned b128 loads. Wa1(64x64)+relu staged in LDS, Wa2(64x32), softmax over
// 32 channels/row, prob written out, value=prob*v scatter-maxed into agg.
// One wave per 16 rows (= 2 edges). Grid exact: E*8/16 row tiles.
// ---------------------------------------------------------------------------
__global__ __launch_bounds__(256) void attn_kernel(
    const bf16* __restrict__ qT, const bf16* __restrict__ kT,
    const bf16* __restrict__ Wa1p, const float* __restrict__ ba1,
    const bf16* __restrict__ Wa2p, const float* __restrict__ ba2,
    const float* __restrict__ vf, const int* __restrict__ rowIdx,
    unsigned* __restrict__ agg, float* __restrict__ probOut) {
  __shared__ bf16 stage[8][16 * 64];
  const int wslot = threadIdx.x >> 5;
  const int gw    = (int)((blockIdx.x * blockDim.x + threadIdx.x) >> 5);
  const int lane  = threadIdx.x & 31;
  const int half  = lane >> 4;
  const int m     = lane & 15;
  const int e8    = gw * 16 + m;          // A row for this lane
  const int e     = e8 >> 3, h = e8 & 7;

  const v8f cz = {0.f, 0.f, 0.f, 0.f, 0.f, 0.f, 0.f, 0.f};

  // ---- hidden = relu(qk @ Wa1 + ba1), 16x64 ----
  const bf16* qrow = qT + (size_t)e * 256 + h * 32;  // c = 0..31
  const bf16* krow = kT + (size_t)e * 256 + h * 32;  // c = 32..63
  v16bf aq = load_a_bf16(qrow, 0, half);
  v16bf ak = load_a_bf16(krow, 0, half);
  v8f c1[4] = {cz, cz, cz, cz};
#pragma unroll
  for (int j = 0; j < 4; ++j) {
    c1[j] = wmma_bf16(aq, load_b_packed(Wa1p, 0, j, 4, lane), c1[j]);
    c1[j] = wmma_bf16(ak, load_b_packed(Wa1p, 1, j, 4, lane), c1[j]);
  }
#pragma unroll
  for (int j = 0; j < 4; ++j) {
    const int n = j * 16 + (lane & 15);
    const float bv = ba1[n];
#pragma unroll
    for (int r = 0; r < 8; ++r) {
      const int mm = r + half * 8;
      stage[wslot][mm * 64 + n] = (bf16)fmaxf(c1[j][r] + bv, 0.f);
    }
  }
  __syncthreads();

  // ---- att = hidden @ Wa2 + ba2, 16x32 ----
  v8f c2[2] = {cz, cz};
  const bf16* srow = &stage[wslot][m * 64];
#pragma unroll
  for (int kb = 0; kb < 64; kb += 32) {
    v16bf a = load_a_bf16(srow, kb, half);
#pragma unroll
    for (int j = 0; j < 2; ++j)
      c2[j] = wmma_bf16(a, load_b_packed(Wa2p, kb >> 5, j, 2, lane), c2[j]);
  }

  // ---- softmax over 32 channels per row, prob out, scatter-max ----
  const float invT = 0.17677669529663687f;  // 1/sqrt(32)
  const int n0 = lane & 15, n1 = 16 + (lane & 15);
  const float b0 = ba2[n0], b1 = ba2[n1];
#pragma unroll
  for (int r = 0; r < 8; ++r) {
    const int mm  = r + half * 8;       // C/D layout row for this VGPR slot
    const int re8 = gw * 16 + mm;
    const int ee  = re8 >> 3, hh = re8 & 7;
    float a0 = (c2[0][r] + b0) * invT;
    float a1 = (c2[1][r] + b1) * invT;
    float mx = fmaxf(a0, a1);
#pragma unroll
    for (int s = 8; s; s >>= 1) mx = fmaxf(mx, __shfl_xor(mx, s, 32));
    float e0 = __expf(a0 - mx), e1 = __expf(a1 - mx);
    float sum = e0 + e1;
#pragma unroll
    for (int s = 8; s; s >>= 1) sum += __shfl_xor(sum, s, 32);
    const float inv = 1.f / sum;
    const float pr0 = e0 * inv, pr1 = e1 * inv;
    probOut[((size_t)ee * 32 + n0) * 8 + hh] = pr0;
    probOut[((size_t)ee * 32 + n1) * 8 + hh] = pr1;
    const float vv0 = pr0 * vf[(size_t)ee * 256 + n0 * 8 + hh];
    const float vv1 = pr1 * vf[(size_t)ee * 256 + n1 * 8 + hh];
    const int node = rowIdx[ee];
    atomicMax(agg + (size_t)node * 256 + n0 * 8 + hh, encf(vv0));
    atomicMax(agg + (size_t)node * 256 + n1 * 8 + hh, encf(vv1));
  }
}

// ---------------------------------------------------------------------------
// small elementwise kernels
// ---------------------------------------------------------------------------
__global__ __launch_bounds__(256) void init_agg_kernel(unsigned* a, int n) {
  const int i = blockIdx.x * 256 + threadIdx.x;
  if (i < n) a[i] = 0x007FFFFFu;  // enc(-inf)
}

__global__ __launch_bounds__(256) void decode_agg_kernel(unsigned* a, int n) {
  const int i = blockIdx.x * 256 + threadIdx.x;
  if (i < n) {
    const unsigned e = a[i];
    unsigned u = (e & 0x80000000u) ? (e & 0x7fffffffu) : ~e;
    if (((u >> 23) & 0xffu) == 0xffu) u = 0u;  // non-finite -> 0.0f
    a[i] = u;  // now holds fp32 bit pattern
  }
}

// ---------------------------------------------------------------------------
// host orchestration
// ---------------------------------------------------------------------------
extern "C" void kernel_launch(void* const* d_in, const int* in_sizes, int n_in,
                              void* d_out, int out_size, void* d_ws,
                              size_t ws_size, hipStream_t stream) {
  (void)in_sizes; (void)n_in; (void)out_size; (void)ws_size;

  const float* x0  = (const float*)d_in[0];
  const float* ef0 = (const float*)d_in[1];
  const int*   ei  = (const int*)d_in[2];
  const int* rowI = ei;        // edge_index[0]
  const int* colI = ei + EE;   // edge_index[1]

  // jax pytree flattens the per-layer dict with sorted keys:
  enum { P_Wa1 = 0, P_Wa2, P_We1, P_We2, P_Wk, P_Wq, P_Wu1, P_Wu2, P_Wv,
         P_ba1, P_ba2, P_be1, P_be2, P_bk, P_bq, P_bu1, P_bu2, P_bv };

  float* outX    = (float*)d_out;
  float* outEF   = outX + NN * 256;
  float* outProb = outEF + EE * 256;

  char* ws = (char*)d_ws;
  size_t off = 0;
  auto take = [&](size_t bytes) -> void* {
    void* p = ws + off;
    off = (off + bytes + 255) & ~(size_t)255;
    return p;
  };
  bf16*  h1   = (bf16*)take(EE * 512 * 2);
  bf16*  qb   = (bf16*)take(EE * 256 * 2);   // qT[e][h][c]
  bf16*  kbuf = (bf16*)take(EE * 256 * 2);   // kT[e][h][c]
  float* vf   = (float*)take(EE * 256 * 4);
  float* wEF  = (float*)take(EE * 256 * 4);
  float* wX   = (float*)take(NN * 256 * 4);
  unsigned* agg = (unsigned*)take(NN * 256 * 4);
  bf16*  h2   = (bf16*)take(NN * 512 * 2);
  bf16* wWe1 = (bf16*)take(768 * 512 * 2);
  bf16* wWe2 = (bf16*)take(512 * 256 * 2);
  bf16* wWq  = (bf16*)take(256 * 256 * 2);
  bf16* wWk  = (bf16*)take(256 * 256 * 2);
  bf16* wWv  = (bf16*)take(256 * 256 * 2);
  bf16* wWa1 = (bf16*)take(64 * 64 * 2);
  bf16* wWa2 = (bf16*)take(64 * 32 * 2);
  bf16* wWu1 = (bf16*)take(512 * 512 * 2);
  bf16* wWu2 = (bf16*)take(512 * 256 * 2);

  auto pack = [&](const float* s, bf16* d, int K, int N) {
    repack_w_kernel<<<(K * N + 255) / 256, 256, 0, stream>>>(s, d, K, N);
  };

  const int Emt = (int)(EE / 16);  // 12500 edge row-tiles
  const int Nmt = (int)(NN / 16);  // 1250 node row-tiles

  for (int l = 0; l < 2; ++l) {
    const float* const* pp = (const float* const*)(d_in + 3 + l * 18);
    pack(pp[P_We1], wWe1, 768, 512);
    pack(pp[P_We2], wWe2, 512, 256);
    pack(pp[P_Wq],  wWq,  256, 256);
    pack(pp[P_Wk],  wWk,  256, 256);
    pack(pp[P_Wv],  wWv,  256, 256);
    pack(pp[P_Wa1], wWa1, 64, 64);
    pack(pp[P_Wa2], wWa2, 64, 32);
    pack(pp[P_Wu1], wWu1, 512, 512);
    pack(pp[P_Wu2], wWu2, 512, 256);

    const float* xin  = l ? wX  : x0;
    const float* efin = l ? wEF : ef0;
    float* efout = l ? outEF : wEF;
    float* xout  = l ? outX  : wX;
    float* prb   = outProb + (size_t)l * EE * 256;

    // --- edge MLP: h1 = relu([xi,ef,xj] @ We1 + be1) ---
    gemm_a3_kernel<OM_BF16, true><<<Emt, 256, 0, stream>>>(
        xin, rowI, 256, efin, nullptr, 256, xin, colI, 256,
        Emt, 512, 768, wWe1, pp[P_be1], nullptr, h1);
    // --- edge_out = h1 @ We2 + be2 (relu between layers only) ---
    if (l == 0)
      gemm_bf16a_kernel<true><<<Emt / 2, 256, 0, stream>>>(
          h1, Emt, 256, 512, wWe2, pp[P_be2], efout);
    else
      gemm_bf16a_kernel<false><<<Emt / 2, 256, 0, stream>>>(
          h1, Emt, 256, 512, wWe2, pp[P_be2], efout);

    // --- q/k/v projections (q/k stored transposed for attention) ---
    gemm_a3_kernel<OM_QKT, false><<<Emt / 2, 256, 0, stream>>>(
        xin, rowI, 256, nullptr, nullptr, 0, nullptr, nullptr, 0,
        Emt, 256, 256, wWq, pp[P_bq], nullptr, qb);
    gemm_a3_kernel<OM_QKT, false><<<Emt / 2, 256, 0, stream>>>(
        efin, nullptr, 256, nullptr, nullptr, 0, nullptr, nullptr, 0,
        Emt, 256, 256, wWk, pp[P_bk], nullptr, kbuf);
    gemm_a3_kernel<OM_F32, false><<<Emt / 2, 256, 0, stream>>>(
        xin, colI, 256, nullptr, nullptr, 0, nullptr, nullptr, 0,
        Emt, 256, 256, wWv, pp[P_bv], vf, nullptr);

    // --- attention + softmax + scatter-max ---
    const int aggN = (int)(NN * 256);
    init_agg_kernel<<<(aggN + 255) / 256, 256, 0, stream>>>(agg, aggN);
    attn_kernel<<<(int)(EE * 8 / 16 / 8), 256, 0, stream>>>(
        qb, kbuf, wWa1, pp[P_ba1], wWa2, pp[P_ba2], vf, rowI, agg, prb);
    decode_agg_kernel<<<(aggN + 255) / 256, 256, 0, stream>>>(agg, aggN);

    // --- node update: h2 = relu([x, agg] @ Wu1 + bu1); out = h2 @ Wu2 + bu2 ---
    gemm_a3_kernel<OM_BF16, true><<<Nmt, 256, 0, stream>>>(
        xin, nullptr, 256, (const float*)agg, nullptr, 256, nullptr, nullptr, 0,
        Nmt, 512, 512, wWu1, pp[P_bu1], nullptr, h2);
    if (l == 0)
      gemm_bf16a_kernel<true><<<Nmt / 2, 256, 0, stream>>>(
          h2, Nmt, 256, 512, wWu2, pp[P_bu2], xout);
    else
      gemm_bf16a_kernel<false><<<Nmt / 2, 256, 0, stream>>>(
          h2, Nmt, 256, 512, wWu2, pp[P_bu2], xout);
  }
}
